// MoEMLP_9534827397153
// MI455X (gfx1250) — matrix operand order repaired and checked by reading, not compiled, not run on hardware
//
#include <hip/hip_runtime.h>

// ---------------- problem constants (B=2,S=1024 -> T=2048 tokens) -----------
#define TOK   2048
#define HDIM  1024
#define IDIM  4096
#define NEXP  8
#define NROWS (2 * TOK)   // top-2: every token appears in exactly 2 expert lists

typedef __bf16 v16bf __attribute__((ext_vector_type(16)));
typedef float  v8f   __attribute__((ext_vector_type(8)));
typedef unsigned int u32x4 __attribute__((ext_vector_type(4)));
typedef int v4i __attribute__((vector_size(16)));   // matches async-LDS builtin

#define AS1 __attribute__((address_space(1)))
#define AS3 __attribute__((address_space(3)))

#if __has_builtin(__builtin_amdgcn_global_load_async_to_lds_b128) && \
    __has_builtin(__builtin_amdgcn_s_wait_asynccnt)
#define HAVE_ASYNC 1
#else
#define HAVE_ASYNC 0
#endif

union Frag { u32x4 q[2]; v16bf v; };
union Pk4  { __bf16 b[4]; unsigned long long u; };

__device__ __forceinline__ v8f wmma_bf16(v16bf a, v16bf b, v8f c) {
  return __builtin_amdgcn_wmma_f32_16x16x32_bf16(false, a, false, b,
                                                 (short)0, c, false, false);
}

__device__ __forceinline__ float fast_rcp(float x) {
#if __has_builtin(__builtin_amdgcn_rcpf)
  return __builtin_amdgcn_rcpf(x);
#else
  return 1.f / x;
#endif
}

#if HAVE_ASYNC
__device__ __forceinline__ void async_b128(const void* gsrc, void* ldst) {
  __builtin_amdgcn_global_load_async_to_lds_b128(
      (AS1 v4i*)gsrc, (AS3 v4i*)ldst, 0, 0);
}
#endif

// ---------------------------------------------------------------------------
// Kernel 1: router. One wave per token: logits = x[t,:] @ gate, top-2 softmax.
// ---------------------------------------------------------------------------
__global__ __launch_bounds__(256) void moe_router(
    const float* __restrict__ x, const float* __restrict__ gate,
    int2* __restrict__ eid, float2* __restrict__ wts) {
  const int wave = threadIdx.x >> 5;
  const int lane = threadIdx.x & 31;
  const int t = blockIdx.x * 8 + wave;
  float acc[NEXP];
#pragma unroll
  for (int e = 0; e < NEXP; ++e) acc[e] = 0.f;
  const float* xr = x + (size_t)t * HDIM;
  for (int h = lane; h < HDIM; h += 32) {
    float xv = xr[h];
    const float4* g4 = (const float4*)(gate + h * NEXP);
    float4 g0 = g4[0], g1 = g4[1];
    acc[0] += xv * g0.x; acc[1] += xv * g0.y;
    acc[2] += xv * g0.z; acc[3] += xv * g0.w;
    acc[4] += xv * g1.x; acc[5] += xv * g1.y;
    acc[6] += xv * g1.z; acc[7] += xv * g1.w;
  }
#pragma unroll
  for (int e = 0; e < NEXP; ++e)
#pragma unroll
    for (int d = 16; d >= 1; d >>= 1) acc[e] += __shfl_xor(acc[e], d);
  if (lane == 0) {
    int e0 = 0; float l0 = acc[0];
#pragma unroll
    for (int e = 1; e < NEXP; ++e) if (acc[e] > l0) { l0 = acc[e]; e0 = e; }
    int e1 = (e0 == 0) ? 1 : 0; float l1 = acc[e1];
#pragma unroll
    for (int e = 0; e < NEXP; ++e)
      if (e != e0 && acc[e] > l1) { l1 = acc[e]; e1 = e; }
    float w0 = 1.f / (1.f + __expf(l1 - l0));
    eid[t] = make_int2(e0, e1);
    wts[t] = make_float2(w0, 1.f - w0);
  }
}

// ---------------------------------------------------------------------------
// Kernel 2: deterministic single-wave scan -> per-expert compacted row lists.
// ---------------------------------------------------------------------------
__global__ void moe_scan(const int2* __restrict__ eid,
                         int2* __restrict__ row_of, int* __restrict__ rowtok,
                         int* __restrict__ counts, int* __restrict__ offs) {
  const int lane = threadIdx.x;          // 32 lanes, 64 tokens each
  const int CH = TOK / 32;
  const int t0 = lane * CH;
  int cnt[NEXP];
#pragma unroll
  for (int e = 0; e < NEXP; ++e) cnt[e] = 0;
  for (int i = 0; i < CH; ++i) {
    int2 p = eid[t0 + i];
    cnt[p.x]++; cnt[p.y]++;
  }
  int pre[NEXP], tot[NEXP];
#pragma unroll
  for (int e = 0; e < NEXP; ++e) {
    int v = cnt[e];
#pragma unroll
    for (int d = 1; d < 32; d <<= 1) {
      int n = __shfl_up(v, d);
      if (lane >= d) v += n;
    }
    pre[e] = v - cnt[e];
    tot[e] = __shfl(v, 31);
  }
  int off[NEXP]; int run = 0;
#pragma unroll
  for (int e = 0; e < NEXP; ++e) { off[e] = run; run += tot[e]; }
  if (lane == 0) {
#pragma unroll
    for (int e = 0; e < NEXP; ++e) { counts[e] = tot[e]; offs[e] = off[e]; }
  }
  int base[NEXP];
#pragma unroll
  for (int e = 0; e < NEXP; ++e) base[e] = off[e] + pre[e];
  for (int i = 0; i < CH; ++i) {
    int t = t0 + i;
    int2 p = eid[t];
    int r0 = base[p.x]++;
    int r1 = base[p.y]++;
    row_of[t] = make_int2(r0, r1);
    rowtok[r0] = t;
    rowtok[r1] = t;
  }
}

// ---------------------------------------------------------------------------
// Kernel 3: gather selected rows of x, cast f32 -> bf16.
// ---------------------------------------------------------------------------
__global__ __launch_bounds__(256) void moe_gather(
    const float* __restrict__ x, const int* __restrict__ rowtok,
    __bf16* __restrict__ xg) {
  const int r = blockIdx.x;
  const int t = rowtok[r];
  const int h = threadIdx.x * 4;
  float4 v = *(const float4*)(x + (size_t)t * HDIM + h);
  Pk4 o;
  o.b[0] = (__bf16)v.x; o.b[1] = (__bf16)v.y;
  o.b[2] = (__bf16)v.z; o.b[3] = (__bf16)v.w;
  *(unsigned long long*)(xg + (size_t)r * HDIM + h) = o.u;
}

// ---------------------------------------------------------------------------
// Kernel 4: grouped GEMM1, fused W1/W3 + SiLU.  C[128x64] x2 per block.
//   A tile staged via GLOBAL_LOAD_ASYNC_TO_LDS (bf16, full tiles) or guarded
//   sync path (edge tiles). B = w1/w3 f32 streamed once from HBM, converted
//   to bf16 and stored transposed in LDS so fragments are contiguous reads.
// ---------------------------------------------------------------------------
__global__ __launch_bounds__(256) void moe_gemm1(
    const __bf16* __restrict__ xg, const float* __restrict__ w1,
    const float* __restrict__ w3, const int* __restrict__ counts,
    const int* __restrict__ offs, __bf16* __restrict__ hid) {
  const int e = blockIdx.z;
  const int cnt = counts[e];
  const int m0 = blockIdx.y * 128;
  if (m0 >= cnt) return;
  const int roff = offs[e];
  const int n0 = blockIdx.x * 64;

  __shared__ __bf16 At[128][32];     // A tile, row-major [m][k]
  __shared__ __bf16 B1t[64][32];     // B tiles transposed: [n][k]
  __shared__ __bf16 B3t[64][32];

  const int tid = threadIdx.x;
  const int lane = tid & 31, wave = tid >> 5;
  const int wr = wave >> 1, wc = wave & 1;      // 4x2 wave grid, 32x32 each
  const int lm = lane & 15, kh = lane >> 4;

  v8f accG[2][2], accU[2][2];
#pragma unroll
  for (int i = 0; i < 2; ++i)
#pragma unroll
    for (int j = 0; j < 2; ++j) { accG[i][j] = (v8f)0.f; accU[i][j] = (v8f)0.f; }

  const float* w1e = w1 + (size_t)e * HDIM * IDIM;
  const float* w3e = w3 + (size_t)e * HDIM * IDIM;

  const int arow = tid >> 1, aseg = tid & 1;
  const __bf16* agsrc =
      xg + (size_t)(roff + m0 + arow) * HDIM + aseg * 16;   // + k0 per iter
  const bool fullA = (m0 + 128 <= cnt);

  for (int k0 = 0; k0 < HDIM; k0 += 32) {
    // ---- stage A: 128 rows x 32 k (bf16) ----
#if HAVE_ASYNC
    if (fullA) {
      const __bf16* g = agsrc + k0;
      __bf16* l = &At[arow][aseg * 16];
      async_b128(g, l);
      async_b128(g + 8, l + 8);
    } else
#endif
    {
      u32x4 d0 = (u32x4)0u, d1 = (u32x4)0u;
      if (m0 + arow < cnt) {
        const u32x4* p = (const u32x4*)(agsrc + k0);
        d0 = p[0]; d1 = p[1];
      }
      u32x4* q = (u32x4*)&At[arow][aseg * 16];
      q[0] = d0; q[1] = d1;
    }
    // ---- stage B1/B3: 32k x 64n f32 -> bf16, transposed into LDS ----
#pragma unroll
    for (int pass = 0; pass < 2; ++pass) {
      const int k = pass * 16 + (tid >> 4);
      const int ns = (tid & 15) * 4;
      float4 a = *(const float4*)(w1e + (size_t)(k0 + k) * IDIM + n0 + ns);
      float4 b = *(const float4*)(w3e + (size_t)(k0 + k) * IDIM + n0 + ns);
      B1t[ns + 0][k] = (__bf16)a.x; B1t[ns + 1][k] = (__bf16)a.y;
      B1t[ns + 2][k] = (__bf16)a.z; B1t[ns + 3][k] = (__bf16)a.w;
      B3t[ns + 0][k] = (__bf16)b.x; B3t[ns + 1][k] = (__bf16)b.y;
      B3t[ns + 2][k] = (__bf16)b.z; B3t[ns + 3][k] = (__bf16)b.w;
    }
    // prefetch next K-tile of the weight stream (critical HBM traffic)
    if (k0 + 32 < HDIM) {
      const size_t pk = (size_t)(k0 + 32 + (tid >> 4)) * IDIM + n0 + (tid & 15) * 4;
      __builtin_prefetch(w1e + pk, 0, 0);
      __builtin_prefetch(w3e + pk, 0, 0);
      __builtin_prefetch(w1e + pk + (size_t)16 * IDIM, 0, 0);
      __builtin_prefetch(w3e + pk + (size_t)16 * IDIM, 0, 0);
    }
#if HAVE_ASYNC
    __builtin_amdgcn_s_wait_asynccnt(0);
#endif
    __syncthreads();

    v16bf af[2], b1f[2], b3f[2];
#pragma unroll
    for (int tm = 0; tm < 2; ++tm) {
      const int M = wr * 32 + tm * 16 + lm;
      Frag f;
      f.q[0] = *(const u32x4*)&At[M][kh * 16];
      f.q[1] = *(const u32x4*)&At[M][kh * 16 + 8];
      af[tm] = f.v;
    }
#pragma unroll
    for (int tn = 0; tn < 2; ++tn) {
      const int N = wc * 32 + tn * 16 + lm;
      Frag f, g;
      f.q[0] = *(const u32x4*)&B1t[N][kh * 16];
      f.q[1] = *(const u32x4*)&B1t[N][kh * 16 + 8];
      b1f[tn] = f.v;
      g.q[0] = *(const u32x4*)&B3t[N][kh * 16];
      g.q[1] = *(const u32x4*)&B3t[N][kh * 16 + 8];
      b3f[tn] = g.v;
    }
#pragma unroll
    for (int tm = 0; tm < 2; ++tm)
#pragma unroll
      for (int tn = 0; tn < 2; ++tn) {
        accG[tm][tn] = wmma_bf16(af[tm], b1f[tn], accG[tm][tn]);
        accU[tm][tn] = wmma_bf16(af[tm], b3f[tn], accU[tm][tn]);
      }
    __syncthreads();
  }

  // epilogue: hid = silu(g) * u, cast to bf16
#pragma unroll
  for (int tm = 0; tm < 2; ++tm)
#pragma unroll
    for (int tn = 0; tn < 2; ++tn) {
      const int Mb = wr * 32 + tm * 16 + kh * 8;
      const int N = n0 + wc * 32 + tn * 16 + lm;
#pragma unroll
      for (int v = 0; v < 8; ++v) {
        const int m = m0 + Mb + v;
        if (m < cnt) {
          float g = accG[tm][tn][v], u = accU[tm][tn][v];
          float s = g * fast_rcp(1.f + __expf(-g));
          hid[(size_t)(roff + m) * IDIM + N] = (__bf16)(s * u);
        }
      }
    }
}

// ---------------------------------------------------------------------------
// Kernel 5: grouped GEMM2: eo = hid @ w2 (f32 out), same tiling, K=IDIM.
// ---------------------------------------------------------------------------
__global__ __launch_bounds__(256) void moe_gemm2(
    const __bf16* __restrict__ hid, const float* __restrict__ w2,
    const int* __restrict__ counts, const int* __restrict__ offs,
    float* __restrict__ eo) {
  const int e = blockIdx.z;
  const int cnt = counts[e];
  const int m0 = blockIdx.y * 128;
  if (m0 >= cnt) return;
  const int roff = offs[e];
  const int n0 = blockIdx.x * 64;

  __shared__ __bf16 At[128][32];
  __shared__ __bf16 Bt[64][32];

  const int tid = threadIdx.x;
  const int lane = tid & 31, wave = tid >> 5;
  const int wr = wave >> 1, wc = wave & 1;
  const int lm = lane & 15, kh = lane >> 4;

  v8f acc[2][2];
#pragma unroll
  for (int i = 0; i < 2; ++i)
#pragma unroll
    for (int j = 0; j < 2; ++j) acc[i][j] = (v8f)0.f;

  const float* w2e = w2 + (size_t)e * IDIM * HDIM;

  const int arow = tid >> 1, aseg = tid & 1;
  const __bf16* agsrc =
      hid + (size_t)(roff + m0 + arow) * IDIM + aseg * 16;  // + k0 per iter
  const bool fullA = (m0 + 128 <= cnt);

  for (int k0 = 0; k0 < IDIM; k0 += 32) {
#if HAVE_ASYNC
    if (fullA) {
      const __bf16* g = agsrc + k0;
      __bf16* l = &At[arow][aseg * 16];
      async_b128(g, l);
      async_b128(g + 8, l + 8);
    } else
#endif
    {
      u32x4 d0 = (u32x4)0u, d1 = (u32x4)0u;
      if (m0 + arow < cnt) {
        const u32x4* p = (const u32x4*)(agsrc + k0);
        d0 = p[0]; d1 = p[1];
      }
      u32x4* q = (u32x4*)&At[arow][aseg * 16];
      q[0] = d0; q[1] = d1;
    }
#pragma unroll
    for (int pass = 0; pass < 2; ++pass) {
      const int k = pass * 16 + (tid >> 4);
      const int ns = (tid & 15) * 4;
      float4 a = *(const float4*)(w2e + (size_t)(k0 + k) * HDIM + n0 + ns);
      Bt[ns + 0][k] = (__bf16)a.x; Bt[ns + 1][k] = (__bf16)a.y;
      Bt[ns + 2][k] = (__bf16)a.z; Bt[ns + 3][k] = (__bf16)a.w;
    }
    if (k0 + 32 < IDIM) {
      const size_t pk = (size_t)(k0 + 32 + (tid >> 4)) * HDIM + n0 + (tid & 15) * 4;
      __builtin_prefetch(w2e + pk, 0, 0);
      __builtin_prefetch(w2e + pk + (size_t)16 * HDIM, 0, 0);
    }
#if HAVE_ASYNC
    __builtin_amdgcn_s_wait_asynccnt(0);
#endif
    __syncthreads();

    v16bf af[2], bf[2];
#pragma unroll
    for (int tm = 0; tm < 2; ++tm) {
      const int M = wr * 32 + tm * 16 + lm;
      Frag f;
      f.q[0] = *(const u32x4*)&At[M][kh * 16];
      f.q[1] = *(const u32x4*)&At[M][kh * 16 + 8];
      af[tm] = f.v;
    }
#pragma unroll
    for (int tn = 0; tn < 2; ++tn) {
      const int N = wc * 32 + tn * 16 + lm;
      Frag f;
      f.q[0] = *(const u32x4*)&Bt[N][kh * 16];
      f.q[1] = *(const u32x4*)&Bt[N][kh * 16 + 8];
      bf[tn] = f.v;
    }
#pragma unroll
    for (int tm = 0; tm < 2; ++tm)
#pragma unroll
      for (int tn = 0; tn < 2; ++tn)
        acc[tm][tn] = wmma_bf16(af[tm], bf[tn], acc[tm][tn]);
    __syncthreads();
  }

#pragma unroll
  for (int tm = 0; tm < 2; ++tm)
#pragma unroll
    for (int tn = 0; tn < 2; ++tn) {
      const int Mb = wr * 32 + tm * 16 + kh * 8;
      const int N = n0 + wc * 32 + tn * 16 + lm;
#pragma unroll
      for (int v = 0; v < 8; ++v) {
        const int m = m0 + Mb + v;
        if (m < cnt) eo[(size_t)(roff + m) * HDIM + N] = acc[tm][tn][v];
      }
    }
}

// ---------------------------------------------------------------------------
// Kernel 6: combine: out[t] = w0*eo[r0] + w1*eo[r1]  (deterministic)
// ---------------------------------------------------------------------------
__global__ __launch_bounds__(256) void moe_combine(
    const float* __restrict__ eo, const int2* __restrict__ row_of,
    const float2* __restrict__ wts, float* __restrict__ out) {
  const int t = blockIdx.x;
  const int h = threadIdx.x * 4;
  int2 r = row_of[t];
  float2 w = wts[t];
  float4 a = *(const float4*)(eo + (size_t)r.x * HDIM + h);
  float4 b = *(const float4*)(eo + (size_t)r.y * HDIM + h);
  float4 o;
  o.x = w.x * a.x + w.y * b.x;
  o.y = w.x * a.y + w.y * b.y;
  o.z = w.x * a.z + w.y * b.z;
  o.w = w.x * a.w + w.y * b.w;
  *(float4*)(out + (size_t)t * HDIM + h) = o;
}

// ---------------------------------------------------------------------------
extern "C" void kernel_launch(void* const* d_in, const int* in_sizes, int n_in,
                              void* d_out, int out_size, void* d_ws, size_t ws_size,
                              hipStream_t stream) {
  const float* x    = (const float*)d_in[0];
  const float* gate = (const float*)d_in[1];
  const float* w1   = (const float*)d_in[2];
  const float* w2   = (const float*)d_in[3];
  const float* w3   = (const float*)d_in[4];
  float* out = (float*)d_out;

  // workspace layout (~56.1 MB total)
  char* ws = (char*)d_ws;
  int2*   eid    = (int2*)(ws);                       // 16 KB
  float2* wts    = (float2*)(ws + (16 << 10));        // 16 KB
  int2*   row_of = (int2*)(ws + (32 << 10));          // 16 KB
  int*    rowtok = (int*)(ws + (48 << 10));           // 16 KB
  int*    counts = (int*)(ws + (64 << 10));           // 32 B
  int*    offs   = (int*)(ws + (64 << 10) + 64);      // 32 B
  __bf16* xg  = (__bf16*)(ws + (128 << 10));                                // 8 MB
  __bf16* hid = (__bf16*)(ws + (128 << 10) + (8ull << 20));                 // 32 MB
  float*  eo  = (float*)(ws + (128 << 10) + (40ull << 20));                 // 16 MB

  moe_router<<<TOK / 8, 256, 0, stream>>>(x, gate, eid, wts);
  moe_scan<<<1, 32, 0, stream>>>(eid, row_of, rowtok, counts, offs);
  moe_gather<<<NROWS, 256, 0, stream>>>(x, rowtok, xg);
  moe_gemm1<<<dim3(IDIM / 64, TOK / 128, NEXP), 256, 0, stream>>>(
      xg, w1, w3, counts, offs, hid);
  moe_gemm2<<<dim3(HDIM / 64, TOK / 128, NEXP), 256, 0, stream>>>(
      hid, w2, counts, offs, eo);
  moe_combine<<<TOK, 256, 0, stream>>>(eo, row_of, wts, out);
}
